// ProbSparseSelfAttention_9371618640135
// MI455X (gfx1250) — matrix operand order repaired and checked by reading, not compiled, not run on hardware
//
#include <hip/hip_runtime.h>

// ---------------------------------------------------------------------------
// ProbSparse attention (Informer) on MI455X / gfx1250.
// Reduction: n_top == L -> top-k scatter overwrites every row -> output is
// exactly full multi-head attention + projections (index_sample is dead).
// Pipeline: f32->f16 weight cvt, 3 WMMA projection GEMMs (V transposed),
// flash attention with online softmax (one wave32 per 32-query block),
// WMMA output GEMM. All matrix math via v_wmma_f32_16x16x32_f16.
// ---------------------------------------------------------------------------

typedef __attribute__((ext_vector_type(16))) _Float16 v16h;
typedef __attribute__((ext_vector_type(8)))  _Float16 v8h;
typedef __attribute__((ext_vector_type(8)))  float    v8f;
typedef __attribute__((ext_vector_type(4)))  float    v4f;

#define B_  2
#define L_  2048
#define DM_ 512
#define H_  8
#define DK_ 64
#define MROWS_ (B_ * L_)   // 4096

static __device__ __forceinline__ v8f wmma_f16(v16h a, v16h b, v8f c) {
  // 8 args: (neg_a, A, neg_b, B, c_mod, C, reuse_a, reuse_b)
  return __builtin_amdgcn_wmma_f32_16x16x32_f16(false, a, false, b, (short)0, c,
                                                false, false);
}

static __device__ __forceinline__ void wave_fence() {
  asm volatile("" ::: "memory");
  __builtin_amdgcn_wave_barrier();
  asm volatile("" ::: "memory");
}

// A-matrix (16x32 f16): lane l holds row l%16.
//   halves e=0..7  -> k = k0 + 8*hi + e          (contiguous run)
//   halves e=8..15 -> k = k0 + 16 + 8*hi + (e-8) (contiguous run)
static __device__ __forceinline__ v16h load_a_f32(const float* __restrict__ base,
                                                  int ld, int row, int k0, int hi) {
  const float* p0 = base + (size_t)row * ld + k0 + 8 * hi;
  const float* p1 = base + (size_t)row * ld + k0 + 16 + 8 * hi;
  v4f x0 = *(const v4f*)p0;
  v4f x1 = *(const v4f*)(p0 + 4);
  v4f y0 = *(const v4f*)p1;
  v4f y1 = *(const v4f*)(p1 + 4);
  v16h a;
#pragma unroll
  for (int j = 0; j < 4; ++j) {
    a[j]      = (_Float16)x0[j];
    a[4 + j]  = (_Float16)x1[j];
    a[8 + j]  = (_Float16)y0[j];
    a[12 + j] = (_Float16)y1[j];
  }
  return a;
}

static __device__ __forceinline__ v16h load_a_f16(const _Float16* __restrict__ base,
                                                  int ld, int row, int k0, int hi) {
  const _Float16* p0 = base + (size_t)row * ld + k0 + 8 * hi;
  const _Float16* p1 = base + (size_t)row * ld + k0 + 16 + 8 * hi;
  v8h lo = *(const v8h*)p0;
  v8h hv = *(const v8h*)p1;
  v16h a;
#pragma unroll
  for (int j = 0; j < 8; ++j) { a[j] = lo[j]; a[8 + j] = hv[j]; }
  return a;
}

// B-matrix (32x16 f16): lane l holds column n = l%16,
// halves e=0..7 -> K = k0 + 8*hi + e ; e=8..15 -> K = k0 + 16 + 8*hi + (e-8).
static __device__ __forceinline__ v16h load_b_f16(const _Float16* __restrict__ base,
                                                  int ld, int row, int k0, int hi) {
  const _Float16* p = base + (size_t)row * ld + k0 + 8 * hi;
  v8h lo = *(const v8h*)p;
  v8h hv = *(const v8h*)(p + 16);
  v16h b;
#pragma unroll
  for (int j = 0; j < 8; ++j) { b[j] = lo[j]; b[8 + j] = hv[j]; }
  return b;
}

// ---------------------------------------------------------------------------
__global__ void cvt_f32_f16_kernel(const float* __restrict__ src,
                                   _Float16* __restrict__ dst, int n) {
  int i = blockIdx.x * blockDim.x + threadIdx.x;
  if (i < n) dst[i] = (_Float16)src[i];
}

// ---------------------------------------------------------------------------
// C = (X @ W^T + bias) * scale ; X: [4096,512] f32, W: [512,512] f16.
// VMODE 0: store f16 as [B,H,L,DK]; VMODE 1: store transposed [B,H,DK,L].
// Block = 256 thr (8 waves); wave does 32 rows x 64 cols (two A tiles share
// every B tile -> 8 wmma per 16 vmem loads). Grid (16, 8).
template <int VMODE>
__global__ __launch_bounds__(256, 1) void gemm_proj_kernel(
    const float* __restrict__ X, const _Float16* __restrict__ W16,
    const float* __restrict__ bias, _Float16* __restrict__ dst, float scale) {
  const int lane = threadIdx.x & 31, wv = threadIdx.x >> 5;
  const int hi = lane >> 4, ln = lane & 15;
  const int rowBase = blockIdx.x * 256 + wv * 32;
  const int nBase   = blockIdx.y * 64;

  v8f acc[2][4];
#pragma unroll
  for (int i = 0; i < 2; ++i)
#pragma unroll
    for (int t = 0; t < 4; ++t)
#pragma unroll
      for (int r = 0; r < 8; ++r) acc[i][t][r] = 0.f;

  for (int k0 = 0; k0 < DM_; k0 += 32) {
    v16h a0 = load_a_f32(X, DM_, rowBase + ln, k0, hi);
    v16h a1 = load_a_f32(X, DM_, rowBase + 16 + ln, k0, hi);
#pragma unroll
    for (int t = 0; t < 4; ++t) {
      v16h b = load_b_f16(W16, DM_, nBase + t * 16 + ln, k0, hi);
      acc[0][t] = wmma_f16(a0, b, acc[0][t]);
      acc[1][t] = wmma_f16(a1, b, acc[1][t]);
    }
  }

  // C layout: lane holds col n = ln, rows m = 8*hi + r within each 16-row tile.
#pragma unroll
  for (int t = 0; t < 4; ++t) {
    int n = nBase + t * 16 + ln;
    float bn = bias[n];
    int h = n >> 6, dk = n & 63;
#pragma unroll
    for (int i = 0; i < 2; ++i) {
#pragma unroll
      for (int r = 0; r < 8; ++r) {
        int gm = rowBase + 16 * i + 8 * hi + r;
        int bb = gm >> 11;            // row / L
        int ls = gm & (L_ - 1);
        _Float16 hvv = (_Float16)((acc[i][t][r] + bn) * scale);
        if (VMODE == 0)
          dst[(((size_t)(bb * H_ + h)) * L_ + ls) * DK_ + dk] = hvv;
        else
          dst[(((size_t)(bb * H_ + h)) * DK_ + dk) * L_ + ls] = hvv;
      }
    }
  }
}

// ---------------------------------------------------------------------------
// Flash-style attention. One wave per 32-query block; 64 steps of 32 keys.
// q16,k16: [B,H,L,DK] f16 (q pre-scaled by 1/sqrt(DK)); vT16: [B,H,DK,L] f16.
// ctx16 out: [B, L, DM] f16. Softmax row pass: one row per lane (32 rows),
// streamed through LDS in two passes so no 32-float register arrays stay live
// (avoids the scratch spills of the aq tiles seen with the register version).
__global__ __launch_bounds__(256, 1) void attn_kernel(
    const _Float16* __restrict__ q16, const _Float16* __restrict__ k16,
    const _Float16* __restrict__ vT16, _Float16* __restrict__ ctx16) {
  __shared__ float Sbuf[8][32][32];
  __shared__ float rowStat[8][32];

  const int lane = threadIdx.x & 31, wv = threadIdx.x >> 5;
  const int hi = lane >> 4, ln = lane & 15;
  const int w  = blockIdx.x * 8 + wv;      // global wave id
  const int bh = w >> 6;                   // 64 q-blocks of 32 rows per (b,h)
  const int qb = w & 63;
  const int q0 = qb * 32;
  const int b  = bh >> 3, h = bh & 7;

  const _Float16* qp = q16 + (size_t)bh * L_ * DK_;
  const _Float16* kp = k16 + (size_t)bh * L_ * DK_;
  const _Float16* vp = vT16 + (size_t)bh * DK_ * L_;

  // Query A-tiles: 2 row-tiles x 2 K-steps, reused across all key chunks.
  v16h aq[2][2];
#pragma unroll
  for (int i = 0; i < 2; ++i)
#pragma unroll
    for (int ks = 0; ks < 2; ++ks)
      aq[i][ks] = load_a_f16(qp, DK_, q0 + 16 * i + ln, 32 * ks, hi);

  float runM = -1e30f, runL = 0.f;    // lane owns softmax row (q0 + lane)
  v8f acc[2][4];
#pragma unroll
  for (int i = 0; i < 2; ++i)
#pragma unroll
    for (int t = 0; t < 4; ++t)
#pragma unroll
      for (int r = 0; r < 8; ++r) acc[i][t][r] = 0.f;

  for (int kc = 0; kc < 64; ++kc) {
    const int n0 = kc * 32;
    if (kc < 63) {  // prefetch next K/V chunk (global_prefetch_b8)
      __builtin_prefetch(kp + (size_t)(n0 + 32) * DK_, 0, 3);
      __builtin_prefetch(vp + (size_t)(n0 + 32), 0, 3);
    }

    // S = q . k^T : 2 q-tiles x 2 n-tiles, K-dim 64 = 2 wmma each (8 wmma).
    v8f s[2][2];
#pragma unroll
    for (int i = 0; i < 2; ++i)
#pragma unroll
      for (int j = 0; j < 2; ++j)
#pragma unroll
        for (int r = 0; r < 8; ++r) s[i][j][r] = 0.f;
#pragma unroll
    for (int j = 0; j < 2; ++j) {
      v16h kb0 = load_b_f16(kp, DK_, n0 + j * 16 + ln, 0, hi);
      v16h kb1 = load_b_f16(kp, DK_, n0 + j * 16 + ln, 32, hi);
#pragma unroll
      for (int i = 0; i < 2; ++i) {
        s[i][j] = wmma_f16(aq[i][0], kb0, s[i][j]);
        s[i][j] = wmma_f16(aq[i][1], kb1, s[i][j]);
      }
    }

    // Spill C-layout S tiles to LDS (per-wave slice; LDS in-order in-wave).
#pragma unroll
    for (int i = 0; i < 2; ++i)
#pragma unroll
      for (int j = 0; j < 2; ++j)
#pragma unroll
        for (int r = 0; r < 8; ++r)
          Sbuf[wv][16 * i + 8 * hi + r][16 * j + ln] = s[i][j][r];
    wave_fence();

    // Row pass, lane owns row (q0 + lane). Pass 1: streaming max (only a
    // v4f live at a time -> no big register arrays).
    float cmax = -1e30f;
#pragma unroll
    for (int c4 = 0; c4 < 8; ++c4) {
      v4f tv = *(const v4f*)&Sbuf[wv][lane][c4 * 4];
      cmax = fmaxf(cmax, fmaxf(fmaxf(tv[0], tv[1]), fmaxf(tv[2], tv[3])));
    }
    float newM = fmaxf(runM, cmax);

    // Pass 2: exp, write P back into Sbuf, accumulate row sum.
    float csum = 0.f;
#pragma unroll
    for (int c4 = 0; c4 < 8; ++c4) {
      v4f tv = *(const v4f*)&Sbuf[wv][lane][c4 * 4];
      v4f ev;
#pragma unroll
      for (int j = 0; j < 4; ++j) ev[j] = __expf(tv[j] - newM);
      *(v4f*)&Sbuf[wv][lane][c4 * 4] = ev;
      csum += (ev[0] + ev[1]) + (ev[2] + ev[3]);
    }
    float alpha = __expf(runM - newM);
    runL = runL * alpha + csum;
    runM = newM;
    rowStat[wv][lane] = alpha;
    wave_fence();

    // Rescale accumulators: acc[i] rows are (16*i + 8*hi + r).
#pragma unroll
    for (int i = 0; i < 2; ++i) {
      v4f a0 = *(const v4f*)&rowStat[wv][16 * i + 8 * hi];
      v4f a1 = *(const v4f*)&rowStat[wv][16 * i + 8 * hi + 4];
      float ar[8] = {a0[0], a0[1], a0[2], a0[3], a1[0], a1[1], a1[2], a1[3]};
#pragma unroll
      for (int t = 0; t < 4; ++t)
#pragma unroll
        for (int r = 0; r < 8; ++r) acc[i][t][r] *= ar[r];
    }

    // Build P A-tiles from Sbuf rows (l%16) and (16 + l%16):
    // halves e=0..7 -> col 8*hi+e ; e=8..15 -> col 16+8*hi+(e-8).
    v16h pm[2];
#pragma unroll
    for (int i = 0; i < 2; ++i) {
      v4f p00 = *(const v4f*)&Sbuf[wv][16 * i + ln][8 * hi];
      v4f p01 = *(const v4f*)&Sbuf[wv][16 * i + ln][8 * hi + 4];
      v4f p10 = *(const v4f*)&Sbuf[wv][16 * i + ln][16 + 8 * hi];
      v4f p11 = *(const v4f*)&Sbuf[wv][16 * i + ln][16 + 8 * hi + 4];
#pragma unroll
      for (int j = 0; j < 4; ++j) {
        pm[i][j]      = (_Float16)p00[j];
        pm[i][4 + j]  = (_Float16)p01[j];
        pm[i][8 + j]  = (_Float16)p10[j];
        pm[i][12 + j] = (_Float16)p11[j];
      }
    }

    // ctx += P (32x32) x V (32x64); every V tile feeds two wmma (8 wmma).
#pragma unroll
    for (int t = 0; t < 4; ++t) {
      v16h vb = load_b_f16(vp, L_, t * 16 + ln, n0, hi);
      acc[0][t] = wmma_f16(pm[0], vb, acc[0][t]);
      acc[1][t] = wmma_f16(pm[1], vb, acc[1][t]);
    }
    wave_fence();
  }

  rowStat[wv][lane] = 1.f / runL;
  wave_fence();

#pragma unroll
  for (int i = 0; i < 2; ++i) {
    v4f l0 = *(const v4f*)&rowStat[wv][16 * i + 8 * hi];
    v4f l1 = *(const v4f*)&rowStat[wv][16 * i + 8 * hi + 4];
    float lr[8] = {l0[0], l0[1], l0[2], l0[3], l1[0], l1[1], l1[2], l1[3]};
#pragma unroll
    for (int t = 0; t < 4; ++t) {
      int col = h * DK_ + t * 16 + ln;
#pragma unroll
      for (int r = 0; r < 8; ++r) {
        int row = q0 + 16 * i + 8 * hi + r;
        ctx16[((size_t)(b * L_ + row)) * DM_ + col] =
            (_Float16)(acc[i][t][r] * lr[r]);
      }
    }
  }
}

// ---------------------------------------------------------------------------
// out = ctx @ Wo^T + bo ; ctx f16 [4096,512], Wo f16, out f32.
__global__ __launch_bounds__(256, 1) void gemm_out_kernel(
    const _Float16* __restrict__ A16, const _Float16* __restrict__ W16,
    const float* __restrict__ bias, float* __restrict__ out) {
  const int lane = threadIdx.x & 31, wv = threadIdx.x >> 5;
  const int hi = lane >> 4, ln = lane & 15;
  const int rowBase = blockIdx.x * 256 + wv * 32;
  const int nBase   = blockIdx.y * 64;

  v8f acc[2][4];
#pragma unroll
  for (int i = 0; i < 2; ++i)
#pragma unroll
    for (int t = 0; t < 4; ++t)
#pragma unroll
      for (int r = 0; r < 8; ++r) acc[i][t][r] = 0.f;

  for (int k0 = 0; k0 < DM_; k0 += 32) {
    v16h a0 = load_a_f16(A16, DM_, rowBase + ln, k0, hi);
    v16h a1 = load_a_f16(A16, DM_, rowBase + 16 + ln, k0, hi);
#pragma unroll
    for (int t = 0; t < 4; ++t) {
      v16h b = load_b_f16(W16, DM_, nBase + t * 16 + ln, k0, hi);
      acc[0][t] = wmma_f16(a0, b, acc[0][t]);
      acc[1][t] = wmma_f16(a1, b, acc[1][t]);
    }
  }

#pragma unroll
  for (int t = 0; t < 4; ++t) {
    int n = nBase + t * 16 + ln;
    float bn = bias[n];
#pragma unroll
    for (int i = 0; i < 2; ++i)
#pragma unroll
      for (int r = 0; r < 8; ++r) {
        int gm = rowBase + 16 * i + 8 * hi + r;
        out[(size_t)gm * DM_ + n] = acc[i][t][r] + bn;
      }
  }
}

// ---------------------------------------------------------------------------
extern "C" void kernel_launch(void* const* d_in, const int* in_sizes, int n_in,
                              void* d_out, int out_size, void* d_ws, size_t ws_size,
                              hipStream_t stream) {
  const float* Q  = (const float*)d_in[0];
  const float* K  = (const float*)d_in[1];
  const float* V  = (const float*)d_in[2];
  const float* Wq = (const float*)d_in[3];
  const float* bq = (const float*)d_in[4];
  const float* Wk = (const float*)d_in[5];
  const float* bk = (const float*)d_in[6];
  const float* Wv = (const float*)d_in[7];
  const float* bv = (const float*)d_in[8];
  const float* Wo = (const float*)d_in[9];
  const float* bo = (const float*)d_in[10];
  // d_in[11] (index_sample) is dead code: n_top == L makes top-k a permutation.
  float* out = (float*)d_out;

  char* ws = (char*)d_ws;
  const size_t WBYTES = (size_t)DM_ * DM_ * sizeof(_Float16);          // 512 KB
  const size_t QKVB   = (size_t)B_ * H_ * L_ * DK_ * sizeof(_Float16); // 4 MB
  _Float16* wq16  = (_Float16*)(ws);
  _Float16* wk16  = (_Float16*)(ws + WBYTES);
  _Float16* wv16  = (_Float16*)(ws + 2 * WBYTES);
  _Float16* wo16  = (_Float16*)(ws + 3 * WBYTES);
  _Float16* q16   = (_Float16*)(ws + 4 * WBYTES);
  _Float16* k16   = (_Float16*)(ws + 4 * WBYTES + QKVB);
  _Float16* vT16  = (_Float16*)(ws + 4 * WBYTES + 2 * QKVB);
  _Float16* ctx16 = (_Float16*)(ws + 4 * WBYTES + 3 * QKVB);

  const int wn = DM_ * DM_;
  cvt_f32_f16_kernel<<<wn / 256, 256, 0, stream>>>(Wq, wq16, wn);
  cvt_f32_f16_kernel<<<wn / 256, 256, 0, stream>>>(Wk, wk16, wn);
  cvt_f32_f16_kernel<<<wn / 256, 256, 0, stream>>>(Wv, wv16, wn);
  cvt_f32_f16_kernel<<<wn / 256, 256, 0, stream>>>(Wo, wo16, wn);

  dim3 ggrid(MROWS_ / 256, DM_ / 64);  // (16, 8)
  gemm_proj_kernel<0><<<ggrid, 256, 0, stream>>>(Q, wq16, bq, q16, 0.125f);
  gemm_proj_kernel<0><<<ggrid, 256, 0, stream>>>(K, wk16, bk, k16, 1.0f);
  gemm_proj_kernel<1><<<ggrid, 256, 0, stream>>>(V, wv16, bv, vT16, 1.0f);

  // B*H*L/32 = 1024 waves, 8 waves per block.
  attn_kernel<<<128, 256, 0, stream>>>(q16, k16, vT16, ctx16);

  gemm_out_kernel<<<ggrid, 256, 0, stream>>>(ctx16, wo16, bo, out);
}
